// Net_61392262529231
// MI455X (gfx1250) — matrix operand order repaired and checked by reading, not compiled, not run on hardware
//
#include <hip/hip_runtime.h>

typedef __attribute__((ext_vector_type(16))) _Float16 v16h;
typedef __attribute__((ext_vector_type(8)))  float    v8f;

#define BB    8
#define TT    256
#define IDIM  80
#define ODIM  80
#define HDIM  256
#define CDIM  64
#define BT    (BB*TT)     /* 2048 */
#define K2    160         /* 2*IDIM */
#define SSH   159         /* shift count: 80 + 2*79 - 80 + 1 */
#define SEN   81          /* encoder offsets: 80 + 160 - 160 + 1 */
#define NITER 4
#define NSWZ  (5*16*32*16)   /* swizzled weight elements = 40960 */

/* ---------------- init: residuals, masks, fragment-swizzled f16 weights --------- */
/* WeSwz layout: fragment-major. For k-step ks, n-tile n, lane L=(lg*16+lm),
   element e (0..15): value = We[(ks*32 + lg*16 + e)*HDIM + n*16 + lm].
   Each lane's 16 halves are contiguous (32B) -> 2x global_load_b128 per frag.   */
__global__ void k_init(const float* __restrict__ x, const float* __restrict__ y,
                       const float* __restrict__ We,
                       float* __restrict__ x_res, float* __restrict__ y_res,
                       unsigned char* __restrict__ smask, int* __restrict__ smaskt,
                       _Float16* __restrict__ WeSwz, float* __restrict__ lossbuf) {
  int gid = blockIdx.x * blockDim.x + threadIdx.x;
  int stride = gridDim.x * blockDim.x;
  for (int i = gid; i < BT * IDIM; i += stride) {
    float xv = x[i], yv = y[i];
    x_res[i] = xv; y_res[i] = yv;
    smask[i] = (yv == 0.f) ? 1 : 0;
  }
  for (int i = gid; i < NSWZ; i += stride) {
    int e  = i & 15;
    int ln = (i >> 4) & 31;
    int n  = (i >> 9) & 15;
    int ks = i >> 13;
    int lm = ln & 15, lg = ln >> 4;
    WeSwz[i] = (_Float16)We[(ks * 32 + lg * 16 + e) * HDIM + n * 16 + lm];
  }
  for (int i = gid; i < BT; i += stride) {
    int all0 = 1;
    for (int d = 0; d < ODIM; ++d) if (y[i * ODIM + d] != 0.f) { all0 = 0; break; }
    smaskt[i] = all0;
  }
  if (gid < NITER) lossbuf[gid] = 0.f;
}

/* ---------------- alignment: sim argmax (+ optional attention/reverse-shift) ---- */
/* mode 0: out = reverse_shift(y_al * softmax(y_al*y_res), theta)  (x branch)
   mode 1: out = y_al                                               (src branch)
   Output written as f16 (consumed only by the WMMA encoder).                      */
__global__ void __launch_bounds__(256) k_align(const float* __restrict__ z,
                                               const float* __restrict__ yres,
                                               _Float16* __restrict__ outv,
                                               int* __restrict__ theta_out, int mode) {
  int bt = blockIdx.x, tid = threadIdx.x;
  __shared__ float xp[240];      /* z padded by 79 each side (len 238, rounded) */
  __shared__ float yv[ODIM];
  __shared__ float yal[ODIM];
  __shared__ float prod[ODIM];
  __shared__ float sval[256];
  __shared__ int   sidx[256];

  for (int i = tid; i < 240; i += 256) {
    int j = i - 79;
    xp[i] = (j >= 0 && j < IDIM) ? z[bt * IDIM + j] : 0.f;
  }
  for (int i = tid; i < ODIM; i += 256) yv[i] = yres[bt * ODIM + i];
  __syncthreads();

  float ny2 = 0.f;
  for (int w = 0; w < ODIM; ++w) ny2 += yv[w] * yv[w];
  float ny = sqrtf(ny2);

  float simv = -1e30f;
  if (tid < SSH) {
    float num = 0.f, ss = 0.f;
    for (int w = 0; w < ODIM; ++w) { float a = xp[tid + w]; num += a * yv[w]; ss += a * a; }
    simv = num / (sqrtf(ss) * ny + 1e-6f);
  }
  sval[tid] = simv; sidx[tid] = tid;
  __syncthreads();
  for (int off = 128; off > 0; off >>= 1) {
    if (tid < off) {
      float v2 = sval[tid + off]; int i2 = sidx[tid + off];
      if (v2 > sval[tid] || (v2 == sval[tid] && i2 < sidx[tid])) { sval[tid] = v2; sidx[tid] = i2; }
    }
    __syncthreads();
  }
  int theta = sidx[0];
  if (tid == 0) theta_out[bt] = theta;

  if (mode == 1) {
    if (tid < ODIM) outv[bt * ODIM + tid] = (_Float16)xp[theta + tid];
    return;
  }
  if (tid < ODIM) yal[tid] = xp[theta + tid];
  __syncthreads();
  float mx = -1e30f;
  for (int w = 0; w < ODIM; ++w) { float l = yal[w] * yv[w]; mx = l > mx ? l : mx; }
  float sum = 0.f;
  for (int w = 0; w < ODIM; ++w) sum += __expf(yal[w] * yv[w] - mx);
  if (tid < ODIM) {
    float attn = __expf(yal[tid] * yv[tid] - mx) / sum;
    prod[tid] = yal[tid] * attn;
  }
  __syncthreads();
  if (tid < IDIM) {
    int src = 79 - theta + tid;  /* inverse shift placement */
    outv[bt * IDIM + tid] = (_Float16)((src >= 0 && src < ODIM) ? prod[src] : 0.f);
  }
}

/* ---------------- WMMA encoder: h = zw(81x160) @ We(160x256) + be --------------- */
/* One workgroup (8 waves) per (b,t). The 160-byte f16 activation row is DMA'd
   into LDS with GLOBAL_LOAD_ASYNC_TO_LDS_B128 (ASYNCcnt), halo zeroed with ds
   stores. B fragments come pre-swizzled from global (2x b128 per fragment).
   6 M-tiles x 16 N-tiles; wave w owns n-tiles {2w,2w+1} x all m; k fully
   unrolled (5 steps) -> 60 v_wmma per wave. Energy argmax via LDS ds_add_f32. */
__global__ void __launch_bounds__(256) k_encode(const _Float16* __restrict__ z,
                                                const _Float16* __restrict__ WeSwz,
                                                const float* __restrict__ be,
                                                float* __restrict__ hout,
                                                int* __restrict__ indout) {
  int bt = blockIdx.x, tid = threadIdx.x;
  __shared__ __attribute__((aligned(16))) _Float16 zp[256];
  __shared__ float energy[96];
  __shared__ int s_ind;

  /* zero halo: [0,80) and [160,256) */
  if (tid < IDIM) zp[tid] = (_Float16)0.f;
  if (tid >= 160) zp[tid] = (_Float16)0.f;
  /* async DMA of the 160B activation row into zp[80..159] (10 lanes x 16B) */
  if (tid < 10) {
    unsigned ldsoff = (unsigned)(uintptr_t)(&zp[IDIM]) + (unsigned)tid * 16u;
    unsigned voff   = (unsigned)tid * 16u;
    unsigned long long gbase = (unsigned long long)(uintptr_t)(z + (size_t)bt * IDIM);
    asm volatile("global_load_async_to_lds_b128 %0, %1, %2"
                 :: "v"(ldsoff), "v"(voff), "s"(gbase) : "memory");
  }
  asm volatile("s_wait_asynccnt 0" ::: "memory");
  if (tid < 96) energy[tid] = 0.f;
  __syncthreads();

  int lane = tid & 31, wave = tid >> 5;
  int lm = lane & 15, lg = lane >> 4;

  __builtin_prefetch(WeSwz, 0, 1);   /* global_prefetch_b8 */

  const v16h* __restrict__ Bv = (const v16h*)WeSwz;

  v8f acc[12];
  v8f vz = {0.f, 0.f, 0.f, 0.f, 0.f, 0.f, 0.f, 0.f};
#pragma unroll
  for (int t = 0; t < 12; ++t) acc[t] = vz;

#pragma unroll
  for (int ks = 0; ks < 5; ++ks) {
    int k0 = ks * 32;
    /* A fragments (16x32 f16, contiguous layout: K=g*16+2j, matching pre-swizzled B) */
    v16h af[6];
#pragma unroll
    for (int m = 0; m < 6; ++m) {
      int srow = m * 16 + lm;
#pragma unroll
      for (int j = 0; j < 8; ++j) {
        int kk = lg * 16 + j * 2;
        af[m][2 * j]     = zp[srow + k0 + kk];
        af[m][2 * j + 1] = zp[srow + k0 + kk + 1];
      }
    }
    /* B fragments: one contiguous 32B v16h per lane (pre-swizzled) */
    v16h bf[2];
#pragma unroll
    for (int q = 0; q < 2; ++q)
      bf[q] = Bv[(size_t)(ks * 16 + (wave * 2 + q)) * 32 + lane];
#pragma unroll
    for (int t = 0; t < 12; ++t) {
      acc[t] = __builtin_amdgcn_wmma_f32_16x16x32_f16(
          false, af[t % 6], false, bf[t / 6], (short)0, acc[t], false, false);
    }
  }

  /* bias + per-row energy (C layout: lane=N, row = j + 8*lg within tile) */
#pragma unroll
  for (int t = 0; t < 12; ++t) {
    int m = t % 6;
    int ncol = (wave * 2 + t / 6) * 16 + lm;
    float bias = be[ncol];
#pragma unroll
    for (int j = 0; j < 8; ++j) {
      int s = m * 16 + j + 8 * lg;
      float v = acc[t][j] + bias;
      acc[t][j] = v;
      if (s < SEN) atomicAdd(&energy[s], v * v);
    }
  }
  __syncthreads();
  if (tid == 0) {
    float b = -1.f; int bi = 0;
    for (int s = 0; s < SEN; ++s) if (energy[s] > b) { b = energy[s]; bi = s; }
    s_ind = bi; indout[bt] = bi;
  }
  __syncthreads();
  int sel = s_ind;
#pragma unroll
  for (int t = 0; t < 12; ++t) {
    int m = t % 6;
    int ncol = (wave * 2 + t / 6) * 16 + lm;
#pragma unroll
    for (int j = 0; j < 8; ++j) {
      int s = m * 16 + j + 8 * lg;
      if (s == sel) hout[bt * HDIM + ncol] = acc[t][j];
    }
  }
}

/* ---------------- hsr: stable top-64 energy masking + exclusivity loss ---------- */
__global__ void __launch_bounds__(256) k_hsr(float* __restrict__ h, float* __restrict__ mp,
                                             const int* __restrict__ smaskt,
                                             float* __restrict__ loss_slot, int first) {
  int bt = blockIdx.x, n = threadIdx.x;
  __shared__ float e[HDIM];
  __shared__ float red[HDIM];
  float val = h[bt * HDIM + n];
  e[n] = val * val;
  __syncthreads();
  float en = e[n];
  int cnt = 0;
  for (int m = 0; m < HDIM; ++m) { float em = e[m]; cnt += (em > en) || (em == en && m < n); }
  float mc = (cnt < CDIM) ? 1.f : 0.f;
  float contrib = 0.f;
  if (first) {
    h[bt * HDIM + n] = val * mc;
    mp[bt * HDIM + n] = mc;
  } else {
    float mpv = mp[bt * HDIM + n];
    float inter = mpv * mc;
    float dl = val - (1.f - inter);
    float lh = smaskt[bt] ? 0.f : dl * dl;
    contrib = (inter > 0.f) ? lh : 0.f;
    float h2 = (mpv > 0.f) ? 0.f : val;
    __syncthreads();
    e[n] = h2 * h2;
    __syncthreads();
    float en2 = e[n]; int c2 = 0;
    for (int m = 0; m < HDIM; ++m) { float em = e[m]; c2 += (em > en2) || (em == en2 && m < n); }
    float mc2 = (c2 < CDIM) ? 1.f : 0.f;
    h[bt * HDIM + n] = h2 * mc2;
    mp[bt * HDIM + n] = mpv + mc2;
  }
  red[n] = contrib;
  __syncthreads();
  for (int off = 128; off > 0; off >>= 1) {
    if (n < off) red[n] += red[n + off];
    __syncthreads();
  }
  if (n == 0 && !first) atomicAdd(loss_slot, red[0]);
}

/* ---------------- decode (windowed GEMV) + masked MSE loss ---------------------- */
__global__ void __launch_bounds__(128) k_decode(const float* __restrict__ h,
                                                const float* __restrict__ Wdec,
                                                const float* __restrict__ bdec,
                                                const int* __restrict__ ind,
                                                const int* __restrict__ theta,
                                                const float* __restrict__ res,
                                                const unsigned char* __restrict__ smask,
                                                float* __restrict__ dec,
                                                float* __restrict__ loss_slot) {
  int bt = blockIdx.x, tid = threadIdx.x;
  __shared__ float hs[HDIM];
  __shared__ float red[128];
  hs[tid] = h[bt * HDIM + tid];
  hs[tid + 128] = h[bt * HDIM + tid + 128];
  __syncthreads();
  float contrib = 0.f;
  if (tid < IDIM) {
    int col = ind[bt] + tid;              /* in [0,160) */
    float s = bdec[col];
    for (int k = 0; k < HDIM; ++k) s += hs[k] * Wdec[k * K2 + col];
    dec[bt * IDIM + tid] = s;
    int th = theta[bt];
    int move = th - 79; if (move < 0) move = -move;
    if (move <= 40 && !smask[bt * IDIM + tid]) {   /* ETH = 40 */
      float d = s - res[bt * IDIM + tid];
      contrib = d * d / (float)(move + 1);
    }
  }
  red[tid] = contrib;
  __syncthreads();
  for (int off = 64; off > 0; off >>= 1) { if (tid < off) red[tid] += red[tid + off]; __syncthreads(); }
  if (tid == 0) atomicAdd(loss_slot, red[0]);
}

/* ---------------- residual update + final mean ---------------------------------- */
__global__ void k_resupd(float* __restrict__ xr, float* __restrict__ yr,
                         const float* __restrict__ xd, const float* __restrict__ yd) {
  int i = blockIdx.x * blockDim.x + threadIdx.x;
  if (i < BT * IDIM) { xr[i] -= xd[i]; yr[i] -= yd[i]; }
}
__global__ void k_final(const float* __restrict__ lossbuf, float* __restrict__ out) {
  if (blockIdx.x == 0 && threadIdx.x == 0)
    out[0] = (lossbuf[0] + lossbuf[1] + lossbuf[2] + lossbuf[3]) * 0.25f;
}

/* ---------------- host launcher -------------------------------------------------- */
extern "C" void kernel_launch(void* const* d_in, const int* in_sizes, int n_in,
                              void* d_out, int out_size, void* d_ws, size_t ws_size,
                              hipStream_t stream) {
  const float* x   = (const float*)d_in[0];
  const float* y   = (const float*)d_in[1];
  const float* We  = (const float*)d_in[2];
  const float* be  = (const float*)d_in[3];
  const float* Wd  = (const float*)d_in[4];
  const float* bd  = (const float*)d_in[5];
  const float* Wds = (const float*)d_in[6];
  const float* bds = (const float*)d_in[7];
  float* out = (float*)d_out;
  (void)in_sizes; (void)n_in; (void)out_size; (void)ws_size;

  char* base = (char*)d_ws; size_t off = 0;
  auto alloc = [&](size_t bytes) -> void* {
    void* p = base + off;
    off = (off + bytes + 255) & ~(size_t)255;
    return p;
  };
  float* x_res = (float*)alloc((size_t)BT * IDIM * 4);
  float* y_res = (float*)alloc((size_t)BT * IDIM * 4);
  float* x_dec = (float*)alloc((size_t)BT * IDIM * 4);
  float* y_dec = (float*)alloc((size_t)BT * IDIM * 4);
  _Float16* x_tmp = (_Float16*)alloc((size_t)BT * IDIM * 2);
  _Float16* y_al2 = (_Float16*)alloc((size_t)BT * IDIM * 2);
  float* hbuf  = (float*)alloc((size_t)BT * HDIM * 4);
  float* mp_self = (float*)alloc((size_t)BT * HDIM * 4);
  float* mp_src  = (float*)alloc((size_t)BT * HDIM * 4);
  _Float16* WeSwz = (_Float16*)alloc((size_t)NSWZ * 2);
  unsigned char* smask = (unsigned char*)alloc((size_t)BT * IDIM);
  int* smaskt = (int*)alloc((size_t)BT * 4);
  int* th1  = (int*)alloc((size_t)BT * 4);
  int* th2  = (int*)alloc((size_t)BT * 4);
  int* ind1 = (int*)alloc((size_t)BT * 4);
  int* ind2 = (int*)alloc((size_t)BT * 4);
  float* lossbuf = (float*)alloc(NITER * 4);

  k_init<<<640, 256, 0, stream>>>(x, y, We, x_res, y_res, smask, smaskt, WeSwz, lossbuf);
  for (int it = 0; it < NITER; ++it) {
    /* self branch */
    k_align<<<BT, 256, 0, stream>>>(x_res, y_res, x_tmp, th1, 0);
    k_encode<<<BT, 256, 0, stream>>>(x_tmp, WeSwz, be, hbuf, ind1);
    k_hsr<<<BT, 256, 0, stream>>>(hbuf, mp_self, smaskt, lossbuf + it, it == 0 ? 1 : 0);
    k_decode<<<BT, 128, 0, stream>>>(hbuf, Wds, bds, ind1, th1, x_res, smask, x_dec, lossbuf + it);
    /* src branch */
    k_align<<<BT, 256, 0, stream>>>(x_dec, y_res, y_al2, th2, 1);
    k_encode<<<BT, 256, 0, stream>>>(y_al2, WeSwz, be, hbuf, ind2);
    k_hsr<<<BT, 256, 0, stream>>>(hbuf, mp_src, smaskt, lossbuf + it, it == 0 ? 1 : 0);
    k_decode<<<BT, 128, 0, stream>>>(hbuf, Wd, bd, ind2, th2, y_res, smask, y_dec, lossbuf + it);
    /* residual update */
    k_resupd<<<(BT * IDIM + 255) / 256, 256, 0, stream>>>(x_res, y_res, x_dec, y_dec);
  }
  k_final<<<1, 32, 0, stream>>>(lossbuf, out);
}
